// RRNNCell_51092930953468
// MI455X (gfx1250) — compile-verified, matrix operand order checked by reading, not run on hardware
//
#include <hip/hip_runtime.h>
#include <hip/hip_bf16.h>
#include <math.h>

typedef __attribute__((ext_vector_type(16))) __bf16 v16bf;
typedef __attribute__((ext_vector_type(8)))  float  v8f;
typedef __attribute__((ext_vector_type(4)))  __bf16 bf16x4;
typedef __attribute__((ext_vector_type(4)))  int    v4i;

#define LL     2048
#define BBATCH 16
#define MM     (LL * BBATCH)   // 32768
#define KK     1024
#define NN     2048
#define NOUT   1024

#define BM 128
#define BN 128
#define KT 32
#define NKIT (KK / KT)         // 32
#define LDA 40                 // bf16 elems per LDS row (80 B: 16B-aligned, non-pow2 banks)
#define LDB 40

#define NCHAN  (BBATCH * NOUT) // 16384 channels
#define NCHUNK 32
#define CLEN   (LL / NCHUNK)   // 64

#if __has_builtin(__builtin_amdgcn_global_load_async_to_lds_b128)
#define HAVE_ASYNC_B128 1
typedef __attribute__((address_space(1))) void gvoid;
typedef __attribute__((address_space(3))) void lvoid;
typedef __attribute__((address_space(1))) v4i  gv4i;
typedef __attribute__((address_space(3))) v4i  lv4i;
#else
#define HAVE_ASYNC_B128 0
#endif

__device__ __forceinline__ void wait_async0() {
#if __has_builtin(__builtin_amdgcn_s_wait_asynccnt)
    __builtin_amdgcn_s_wait_asynccnt(0);
#else
    asm volatile("s_wait_asynccnt 0x0" ::: "memory");
#endif
}

// ---------------------------------------------------------------------------
// Kernel 1: weight transpose + f32->bf16 convert + gate de-interleave.
// Wt row n' (group g = n'/32, j = n'%32):
//   j <  16 : original column 2*(g*16 + j)       (u0 pre-act, channel g*16+j)
//   j >= 16 : original column 2*(g*16 + j-16)+1  (forget pre-act, same channel)
// u0/forget for the SAME channel land in the SAME lane of adjacent 16-wide
// WMMA output tiles -> gate epilogue needs no cross-lane shuffles.
// ---------------------------------------------------------------------------
__global__ __launch_bounds__(256)
void wt_transpose(const float* __restrict__ W, __bf16* __restrict__ Wt) {
    int idx = blockIdx.x * 256 + threadIdx.x;   // 0 .. NN*KK-1
    int np  = idx >> 10;                        // permuted row (0..2047)
    int k   = idx & (KK - 1);
    int g    = np >> 5;
    int j    = np & 31;
    int chan = (g << 4) | (j & 15);
    int gate = (j >> 4) & 1;
    int col  = (chan << 1) | gate;              // original interleaved column
    Wt[idx] = (__bf16)W[(size_t)k * NN + col];
}

// ---------------------------------------------------------------------------
// Kernel 2: double-buffered bf16 WMMA GEMM + fused, shuffle-free gate epilogue.
// ---------------------------------------------------------------------------
__global__ __launch_bounds__(256)
void gemm_gates(const float*  __restrict__ A,     // [MM,KK] f32
                const __bf16* __restrict__ Wt,    // [NN,KK] bf16 (permuted rows)
                const float*  __restrict__ bias,  // [2048]
                float* __restrict__ u0_out,       // [MM,NOUT] (cs slot)
                float* __restrict__ f_out) {      // [MM,NOUT]
    __shared__ __bf16 Asl[2][BM * LDA];
    __shared__ __bf16 Bsl[2][BN * LDB];

    const int t    = threadIdx.x;
    const int wave = t >> 5;
    const int lane = t & 31;
    const int wm   = wave >> 1;            // 0..3 (M, 32 rows each)
    const int wn   = wave & 1;             // 0..1 (N, 64 cols each)
    const int m0   = blockIdx.y * BM;
    const int n0   = blockIdx.x * BN;
    const int l15   = lane & 15;
    const int khalf = (lane < 16) ? 0 : 8; // 16-bit fragment K split
    const int mhalf = lane >> 4;

    v8f acc[2][4];
    const v8f vz = {0.f, 0.f, 0.f, 0.f, 0.f, 0.f, 0.f, 0.f};
#pragma unroll
    for (int i = 0; i < 2; ++i)
#pragma unroll
        for (int j = 0; j < 4; ++j) acc[i][j] = vz;

    // ---- staging helpers ----
    float4 areg[4];
    auto loadA = [&](int kb) {
#pragma unroll
        for (int i = 0; i < 4; ++i) {
            int idx = t + i * 256;          // 0..1023
            int r = idx >> 3, c4 = idx & 7;
            areg[i] = *(const float4*)(A + (size_t)(m0 + r) * KK + kb + c4 * 4);
        }
    };
    auto storeA = [&](int buf) {
#pragma unroll
        for (int i = 0; i < 4; ++i) {
            int idx = t + i * 256;
            int r = idx >> 3, c4 = idx & 7;
            bf16x4 h;
            h.x = (__bf16)areg[i].x; h.y = (__bf16)areg[i].y;
            h.z = (__bf16)areg[i].z; h.w = (__bf16)areg[i].w;
            *(bf16x4*)(&Asl[buf][r * LDA + c4 * 4]) = h;
        }
    };
#if HAVE_ASYNC_B128
    auto asyncB = [&](int kb, int buf) {
#pragma unroll
        for (int i = 0; i < 2; ++i) {
            int idx = t + i * 256;          // 0..511
            int r = idx >> 2, c16 = idx & 3;
            const __bf16* gp = Wt + (size_t)(n0 + r) * KK + kb + c16 * 8;
            __bf16*       lp = &Bsl[buf][r * LDB + c16 * 8];
            __builtin_amdgcn_global_load_async_to_lds_b128(
                (gv4i*)(gvoid*)(void*)gp, (lv4i*)(lvoid*)lp, 0, 0);
        }
    };
#else
    uint4 breg[2];
    auto loadB = [&](int kb) {
#pragma unroll
        for (int i = 0; i < 2; ++i) {
            int idx = t + i * 256;
            int r = idx >> 2, c16 = idx & 3;
            breg[i] = *(const uint4*)(Wt + (size_t)(n0 + r) * KK + kb + c16 * 8);
        }
    };
    auto storeB = [&](int buf) {
#pragma unroll
        for (int i = 0; i < 2; ++i) {
            int idx = t + i * 256;
            int r = idx >> 2, c16 = idx & 3;
            *(uint4*)(&Bsl[buf][r * LDB + c16 * 8]) = breg[i];
        }
    };
#endif

    // ---- prologue: stage K-tile 0 into buffer 0 ----
    loadA(0);
    storeA(0);
#if HAVE_ASYNC_B128
    asyncB(0, 0);
#else
    loadB(0);
    storeB(0);
#endif

    int cur = 0;
    for (int it = 0; it < NKIT; ++it) {
        const int kb  = it * KT;
        const int nxt = cur ^ 1;
#if HAVE_ASYNC_B128
        wait_async0();                      // my async B copies into buf[cur] done
#endif
        __syncthreads();                    // everyone's tile[cur] ready; prev reads done

        if (it + 1 < NKIT) {                // prefetch next tile
#if HAVE_ASYNC_B128
            asyncB(kb + KT, nxt);
#else
            loadB(kb + KT);
#endif
            loadA(kb + KT);
        }

        // ---- fragments (two ds_load_b128 each) + 8 WMMAs ----
        union F { v16bf v; uint4 q[2]; };
        F a[2], b[4];
#pragma unroll
        for (int mi = 0; mi < 2; ++mi) {
            const __bf16* p = &Asl[cur][(wm * 32 + mi * 16 + l15) * LDA + khalf];
            a[mi].q[0] = *(const uint4*)(p);
            a[mi].q[1] = *(const uint4*)(p + 16);
        }
#pragma unroll
        for (int ni = 0; ni < 4; ++ni) {
            const __bf16* p = &Bsl[cur][(wn * 64 + ni * 16 + l15) * LDB + khalf];
            b[ni].q[0] = *(const uint4*)(p);
            b[ni].q[1] = *(const uint4*)(p + 16);
        }
#pragma unroll
        for (int mi = 0; mi < 2; ++mi)
#pragma unroll
            for (int ni = 0; ni < 4; ++ni)
                acc[mi][ni] = __builtin_amdgcn_wmma_f32_16x16x32_bf16(
                    false, a[mi].v, false, b[ni].v,
                    (short)0, acc[mi][ni], false, false);

        if (it + 1 < NKIT) {                // fill buf[nxt]
            storeA(nxt);
#if !HAVE_ASYNC_B128
            storeB(nxt);
#endif
        }
        cur = nxt;
    }

    // ---- fused gate epilogue (no shuffles: pair tiles 2p / 2p+1) ----
    const int g0 = (n0 + wn * 64) >> 5;     // 32-col permuted group index
#pragma unroll
    for (int p = 0; p < 2; ++p) {
        const int   n   = ((g0 + p) << 4) + l15;     // output channel
        const float fbv = bias[NOUT + n];            // forget bias
#pragma unroll
        for (int mi = 0; mi < 2; ++mi) {
#pragma unroll
            for (int r = 0; r < 8; ++r) {
                float u0pre = acc[mi][2 * p][r];
                float fpre  = acc[mi][2 * p + 1][r] + fbv;
                float f     = __builtin_amdgcn_rcpf(1.0f + __expf(-fpre));
                float u0    = u0pre * (1.0f - f);
                int   m     = m0 + wm * 32 + mi * 16 + mhalf * 8 + r;
                size_t off  = (size_t)m * NOUT + n;
                f_out[off]  = f;
                u0_out[off] = u0;
            }
        }
    }
}

// ---------------------------------------------------------------------------
// Kernels 3/4/5: chunked affine scan  c_t = c_{t-1}*f_t + u_t
// ---------------------------------------------------------------------------
__global__ __launch_bounds__(256)
void scan_passA(const float* __restrict__ u0, const float* __restrict__ fg,
                float* __restrict__ cA, float* __restrict__ cB) {
    int idx = blockIdx.x * 256 + threadIdx.x;   // chunk*NCHAN + cid
    int ch  = idx >> 14;
    int cid = idx & (NCHAN - 1);
    int b   = cid >> 10;
    int n   = cid & (NOUT - 1);
    size_t off = ((size_t)(ch * CLEN) * BBATCH + b) * NOUT + n;
    const size_t stride = (size_t)BBATCH * NOUT;
    float a = 1.f, c = 0.f;
    for (int s = 0; s < CLEN; ++s) {
        float f = fg[off];
        float u = u0[off];
        c = c * f + u;
        a *= f;
        off += stride;
    }
    cA[idx] = a;
    cB[idx] = c;
}

__global__ __launch_bounds__(256)
void scan_passB(const float* __restrict__ cA, const float* __restrict__ cB,
                const float* __restrict__ c_init,
                float* __restrict__ prefix, float* __restrict__ c_final) {
    int cid = blockIdx.x * 256 + threadIdx.x;   // 0..16383
    float p = c_init[cid];
    for (int ch = 0; ch < NCHUNK; ++ch) {
        prefix[ch * NCHAN + cid] = p;
        p = p * cA[ch * NCHAN + cid] + cB[ch * NCHAN + cid];
    }
    c_final[cid] = p;
}

__global__ __launch_bounds__(256)
void scan_passC(float* __restrict__ cs, const float* __restrict__ fg,
                const float* __restrict__ prefix) {
    int idx = blockIdx.x * 256 + threadIdx.x;
    int ch  = idx >> 14;
    int cid = idx & (NCHAN - 1);
    int b   = cid >> 10;
    int n   = cid & (NOUT - 1);
    size_t off = ((size_t)(ch * CLEN) * BBATCH + b) * NOUT + n;
    const size_t stride = (size_t)BBATCH * NOUT;
    float c = prefix[idx];
    for (int s = 0; s < CLEN; ++s) {
        float f = fg[off];
        float u = cs[off];     // cs currently holds u0
        c = c * f + u;
        cs[off] = c;           // overwrite in place with scanned value
        off += stride;
    }
}

// ---------------------------------------------------------------------------
extern "C" void kernel_launch(void* const* d_in, const int* in_sizes, int n_in,
                              void* d_out, int out_size, void* d_ws, size_t ws_size,
                              hipStream_t stream) {
    const float* input  = (const float*)d_in[0];   // [L,B,1024] f32
    const float* c_init = (const float*)d_in[1];   // [B,1024]   f32
    const float* weight = (const float*)d_in[2];   // [1024,2048] f32
    const float* bias   = (const float*)d_in[3];   // [2048]     f32

    float* out     = (float*)d_out;
    float* cs      = out;                                   // [L,B,1024]
    float* c_final = out + (size_t)MM * NOUT;               // [B,1024]
    float* forget  = c_final + NCHAN;                       // [L,B,1024]

    char*   ws     = (char*)d_ws;
    __bf16* Wt     = (__bf16*)ws;                           // 4 MB [NN,KK] bf16
    float*  cA     = (float*)(ws + (size_t)NN * KK * sizeof(__bf16));
    float*  cB     = cA + (size_t)NCHAN * NCHUNK;           // +2 MB
    float*  prefix = cB + (size_t)NCHAN * NCHUNK;           // +2 MB (total 10 MB)

    wt_transpose<<<(NN * KK) / 256, 256, 0, stream>>>(weight, Wt);
    gemm_gates<<<dim3(NN / BN, MM / BM), 256, 0, stream>>>(input, Wt, bias, cs, forget);
    scan_passA<<<(NCHAN * NCHUNK) / 256, 256, 0, stream>>>(cs, forget, cA, cB);
    scan_passB<<<NCHAN / 256, 256, 0, stream>>>(cA, cB, c_init, prefix, c_final);
    scan_passC<<<(NCHAN * NCHUNK) / 256, 256, 0, stream>>>(cs, forget, prefix);
}
